// GCN_31035433681286
// MI455X (gfx1250) — compile-verified, matrix operand order checked by reading, not run on hardware
//
#include <hip/hip_runtime.h>

typedef __attribute__((ext_vector_type(16))) __bf16 v16bf;
typedef __attribute__((ext_vector_type(8)))  __bf16 v8bf;
typedef __attribute__((ext_vector_type(8)))  float  v8f;

#define NN 50000
#define EE 400000
#define GG 64

static __device__ __forceinline__ __bf16 f2bf(float f) {
  unsigned u = __float_as_uint(f);
  u += 0x7fffu + ((u >> 16) & 1u);              // round-to-nearest-even
  unsigned short h = (unsigned short)(u >> 16);
  __bf16 r;
  __builtin_memcpy(&r, &h, sizeof(r));
  return r;
}

// order-preserving float<->uint encoding for atomicMax-based segment_max
static __device__ __forceinline__ unsigned encf(float f) {
  unsigned u = __float_as_uint(f);
  return (u & 0x80000000u) ? ~u : (u | 0x80000000u);
}
static __device__ __forceinline__ float decf(unsigned u) {
  u = (u & 0x80000000u) ? (u & 0x7fffffffu) : ~u;
  return __uint_as_float(u);
}

// ---------------- degree / normalization ----------------
__global__ void k_deg_init(float* deg) {
  int i = blockIdx.x * blockDim.x + threadIdx.x;
  if (i < NN) deg[i] = 1.0f;                      // self-loop
}
__global__ void k_deg_acc(const int* __restrict__ ei, float* deg) {
  int e = blockIdx.x * blockDim.x + threadIdx.x;
  if (e < EE) atomicAdd(&deg[ei[EE + e]], 1.0f);  // dst = ei[E + e]
}
__global__ void k_deg_rsqrt(float* deg) {
  int i = blockIdx.x * blockDim.x + threadIdx.x;
  if (i < NN) deg[i] = rsqrtf(deg[i]);            // dis = deg^-1/2
}

// ---------------- fp32 -> bf16 row-major copy ----------------
__global__ void k_cvt_bf16(const float* __restrict__ in, __bf16* __restrict__ out,
                           int count) {
  int i = blockIdx.x * blockDim.x + threadIdx.x;
  if (i < count) out[i] = f2bf(in[i]);
}

// ---- pack fp32 W[din x dout] into bf16 in WMMA B-fragment order ----
// element idx = ((nt*KT + kt)*32 + lane)*16 + e ; lane: n = nt*16+lane%16,
// half = lane/16 ; e -> K = kt*32 + (e/8)*16 + half*8 + e%8
__global__ void k_pack_w(const float* __restrict__ W, __bf16* __restrict__ Wp,
                         int din, int dout) {
  int idx = blockIdx.x * blockDim.x + threadIdx.x;
  if (idx >= din * dout) return;
  int ktc  = din >> 5;
  int e    = idx & 15;
  int lane = (idx >> 4) & 31;
  int rest = idx >> 9;
  int kt   = rest % ktc;
  int nt   = rest / ktc;
  int n    = (nt << 4) + (lane & 15);
  int half = lane >> 4;
  int k    = (kt << 5) + ((e >> 3) << 4) + (half << 3) + (e & 7);
  Wp[idx]  = f2bf(W[k * dout + n]);
}

// ------- WMMA GEMM: C[rows x DOUT] = A_bf16[rows x DIN] @ W -------
template <int DIN, int DOUT>
__global__ void k_gemm(const __bf16* __restrict__ A, const __bf16* __restrict__ Wp,
                       float* __restrict__ C, int nrows) {
  constexpr int KT = DIN / 32;
  constexpr int NT = DOUT / 16;
  int lane   = threadIdx.x & 31;
  int wv     = threadIdx.x >> 5;
  int base_m = (blockIdx.x * (int)(blockDim.x >> 5) + wv) << 4;
  if (base_m >= nrows) return;                 // wave-uniform exit, EXEC stays full
  int half  = lane >> 4;
  int mlane = lane & 15;
  int mrow  = min(base_m + mlane, nrows - 1);  // clamp, no divergence before WMMA
  bool full = (base_m + 16 <= nrows);          // wave-uniform

  // A fragments for all K-tiles live in VGPRs (<=128 VGPRs for DIN=512)
  v16bf afr[KT];
  const __bf16* arow = A + (size_t)mrow * DIN + (half << 3);
#pragma unroll
  for (int kt = 0; kt < KT; ++kt) {
    v8bf lo = *(const v8bf*)(arow + (kt << 5));        // K = kt*32 + half*8 + 0..7
    v8bf hi = *(const v8bf*)(arow + (kt << 5) + 16);   // K = +16
    afr[kt] = __builtin_shufflevector(lo, hi, 0, 1, 2, 3, 4, 5, 6, 7,
                                      8, 9, 10, 11, 12, 13, 14, 15);
  }

  for (int nt = 0; nt < NT; ++nt) {
    v8f acc = {};
#pragma unroll
    for (int kt = 0; kt < KT; ++kt) {
      v16bf b = *(const v16bf*)(Wp + ((((size_t)nt * KT + kt) << 5) + lane) * 16);
      acc = __builtin_amdgcn_wmma_f32_16x16x32_bf16(false, afr[kt], false, b,
                                                    (short)0, acc, false, false);
    }
    float* crow = C + (size_t)base_m * DOUT + (nt << 4) + mlane;
    if (full) {
#pragma unroll
      for (int r = 0; r < 8; ++r)
        crow[(size_t)(r + (half << 3)) * DOUT] = acc[r];
    } else {
#pragma unroll
      for (int r = 0; r < 8; ++r) {
        int m = base_m + r + (half << 3);
        if (m < nrows) crow[(size_t)(r + (half << 3)) * DOUT] = acc[r];
      }
    }
  }
}

// ---------------- out = ht * self_norm + bias ----------------
template <int DOUT>
__global__ void k_preinit(const float* __restrict__ ht, const float* __restrict__ dis,
                          const float* __restrict__ bias, float* __restrict__ out) {
  int idx = blockIdx.x * blockDim.x + threadIdx.x;
  if (idx >= NN * DOUT) return;
  int node = idx / DOUT;
  int col  = idx - node * DOUT;
  float d = dis[node];
  out[idx] = ht[idx] * (d * d) + bias[col];
}

// -------- edge scatter: out[dst] += ht[src] * dis[src]*dis[dst] --------
template <int DOUT>
__global__ void k_scatter(const int* __restrict__ ei, const float* __restrict__ dis,
                          const float* __restrict__ ht, float* __restrict__ out) {
  constexpr int CP = DOUT / 4;
  int tid = blockIdx.x * blockDim.x + threadIdx.x;
  if (tid >= EE * CP) return;
  int e = tid / CP;
  int c = tid - e * CP;
  int s = ei[e], d = ei[EE + e];
  float nrm = dis[s] * dis[d];
  float4 v = *(const float4*)(ht + (size_t)s * DOUT + (c << 2));
  float* o = out + (size_t)d * DOUT + (c << 2);
  atomicAdd(o + 0, v.x * nrm);
  atomicAdd(o + 1, v.y * nrm);
  atomicAdd(o + 2, v.z * nrm);
  atomicAdd(o + 3, v.w * nrm);
}

// ---- leaky relu in place + bf16 copy for the next WMMA GEMM ----
__global__ void k_lrelu_bf(float* __restrict__ buf, __bf16* __restrict__ obf,
                           int count) {
  int i = blockIdx.x * blockDim.x + threadIdx.x;
  if (i >= count) return;
  float v = buf[i];
  v = v >= 0.f ? v : 0.01f * v;
  buf[i] = v;
  obf[i] = f2bf(v);
}

// ---- LayerNorm(h4 + residual + br) over 64 features, one wave per node ----
__global__ void k_ln_res(const float* __restrict__ h, const float* __restrict__ res,
                         const float* __restrict__ br, const float* __restrict__ g,
                         const float* __restrict__ b, float* __restrict__ out) {
  int node = blockIdx.x * 8 + (int)(threadIdx.x >> 5);
  int lane = threadIdx.x & 31;
  if (node >= NN) return;
  size_t o = (size_t)node * 64;
  float a0 = h[o + lane]      + res[o + lane]      + br[lane];
  float a1 = h[o + 32 + lane] + res[o + 32 + lane] + br[32 + lane];
  float s = a0 + a1, sq = a0 * a0 + a1 * a1;
#pragma unroll
  for (int m = 16; m >= 1; m >>= 1) { s += __shfl_xor(s, m); sq += __shfl_xor(sq, m); }
  float mean = s * (1.f / 64.f);
  float var  = sq * (1.f / 64.f) - mean * mean;
  float inv  = rsqrtf(var + 1e-5f);
  out[o + lane]      = (a0 - mean) * inv * g[lane]      + b[lane];
  out[o + 32 + lane] = (a1 - mean) * inv * g[32 + lane] + b[32 + lane];
}

__global__ void k_pool_zero(unsigned* pool) {
  int i = blockIdx.x * blockDim.x + threadIdx.x;
  if (i < GG * 64) pool[i] = 0u;                 // encodes "most negative"
}
__global__ void k_pool_max(const float* __restrict__ hln, const int* __restrict__ batch,
                           unsigned* pool) {
  int idx = blockIdx.x * blockDim.x + threadIdx.x;
  if (idx >= NN * 64) return;
  int node = idx >> 6, f = idx & 63;
  atomicMax(&pool[batch[node] * 64 + f], encf(hln[idx]));
}

// ---- head: fc1 -> LN -> leaky -> fc2, one thread per graph ----
__global__ void k_head(const unsigned* __restrict__ pool, const float* __restrict__ W1h,
                       const float* __restrict__ b1h, const float* __restrict__ g,
                       const float* __restrict__ bb, const float* __restrict__ W2h,
                       const float* __restrict__ b2h, float* __restrict__ out) {
  int gi = threadIdx.x;
  if (gi >= GG) return;
  float p[64], q[64];
#pragma unroll
  for (int k = 0; k < 64; ++k) p[k] = decf(pool[gi * 64 + k]);
  float s = 0.f, sq = 0.f;
  for (int j = 0; j < 64; ++j) {
    float a = b1h[j];
#pragma unroll
    for (int k = 0; k < 64; ++k) a += p[k] * W1h[k * 64 + j];
    q[j] = a; s += a; sq += a * a;
  }
  float mean = s * (1.f / 64.f);
  float var  = sq * (1.f / 64.f) - mean * mean;
  float inv  = rsqrtf(var + 1e-5f);
#pragma unroll
  for (int j = 0; j < 64; ++j) {
    float v = (q[j] - mean) * inv * g[j] + bb[j];
    q[j] = v >= 0.f ? v : 0.01f * v;
  }
  for (int j = 0; j < 16; ++j) {
    float a = b2h[j];
#pragma unroll
    for (int k = 0; k < 64; ++k) a += q[k] * W2h[k * 16 + j];
    out[gi * 16 + j] = a;
  }
}

extern "C" void kernel_launch(void* const* d_in, const int* in_sizes, int n_in,
                              void* d_out, int out_size, void* d_ws, size_t ws_size,
                              hipStream_t stream) {
  (void)in_sizes; (void)n_in; (void)out_size; (void)ws_size;
  const float* x     = (const float*)d_in[0];
  const int*   ei    = (const int*)d_in[1];
  const int*   batch = (const int*)d_in[2];
  const float* W1 = (const float*)d_in[3];  const float* b1 = (const float*)d_in[4];
  const float* W2 = (const float*)d_in[5];  const float* b2 = (const float*)d_in[6];
  const float* W3 = (const float*)d_in[7];  const float* b3 = (const float*)d_in[8];
  const float* W4 = (const float*)d_in[9];  const float* b4 = (const float*)d_in[10];
  const float* Wr = (const float*)d_in[11]; const float* br = (const float*)d_in[12];
  const float* lng = (const float*)d_in[13]; const float* lnb = (const float*)d_in[14];
  const float* f1W = (const float*)d_in[15]; const float* f1b = (const float*)d_in[16];
  const float* fng = (const float*)d_in[17]; const float* fnb = (const float*)d_in[18];
  const float* f2W = (const float*)d_in[19]; const float* f2b = (const float*)d_in[20];
  float* out = (float*)d_out;

  // workspace carve (256B aligned)
  char* base = (char*)d_ws;
  size_t off = 0;
  auto carve = [&](size_t bytes) -> char* {
    char* p = base + off;
    off += (bytes + 255) & ~(size_t)255;
    return p;
  };
  float*    dis   = (float*)carve((size_t)NN * 4);
  float*    bufA  = (float*)carve((size_t)NN * 512 * 4);   // h (fp32, atomics target)
  float*    bufB  = (float*)carve((size_t)NN * 512 * 4);   // ht
  __bf16*   bufC  = (__bf16*)carve((size_t)NN * 512 * 2);  // h (bf16, GEMM A operand)
  __bf16*   xbf   = (__bf16*)carve((size_t)NN * 128 * 2);
  float*    resid = (float*)carve((size_t)NN * 64 * 4);
  __bf16*   wpack = (__bf16*)carve((size_t)256 * 512 * 2);
  unsigned* poolu = (unsigned*)carve((size_t)GG * 64 * 4);

  const int T = 256;
  auto cdiv = [](long a, long b) { return (int)((a + b - 1) / b); };

  // degree normalization
  k_deg_init<<<cdiv(NN, T), T, 0, stream>>>(dis);
  k_deg_acc<<<cdiv(EE, T), T, 0, stream>>>(ei, dis);
  k_deg_rsqrt<<<cdiv(NN, T), T, 0, stream>>>(dis);

  // x -> bf16 (used by residual GEMM and layer 1)
  k_cvt_bf16<<<cdiv((long)NN * 128, T), T, 0, stream>>>(x, xbf, NN * 128);

  const int gemm_grid = cdiv(NN, 128);  // 8 waves/block * 16 rows/wave

  // residual = x @ Wr   (br folded into LN stage)
  k_pack_w<<<cdiv(128 * 64, T), T, 0, stream>>>(Wr, wpack, 128, 64);
  k_gemm<128, 64><<<gemm_grid, T, 0, stream>>>(xbf, wpack, resid, NN);

  // layer 1: 128 -> 64
  k_pack_w<<<cdiv(128 * 64, T), T, 0, stream>>>(W1, wpack, 128, 64);
  k_gemm<128, 64><<<gemm_grid, T, 0, stream>>>(xbf, wpack, bufB, NN);
  k_preinit<64><<<cdiv((long)NN * 64, T), T, 0, stream>>>(bufB, dis, b1, bufA);
  k_scatter<64><<<cdiv((long)EE * 16, T), T, 0, stream>>>(ei, dis, bufB, bufA);
  k_lrelu_bf<<<cdiv((long)NN * 64, T), T, 0, stream>>>(bufA, bufC, NN * 64);

  // layer 2: 64 -> 256
  k_pack_w<<<cdiv(64 * 256, T), T, 0, stream>>>(W2, wpack, 64, 256);
  k_gemm<64, 256><<<gemm_grid, T, 0, stream>>>(bufC, wpack, bufB, NN);
  k_preinit<256><<<cdiv((long)NN * 256, T), T, 0, stream>>>(bufB, dis, b2, bufA);
  k_scatter<256><<<cdiv((long)EE * 64, T), T, 0, stream>>>(ei, dis, bufB, bufA);
  k_lrelu_bf<<<cdiv((long)NN * 256, T), T, 0, stream>>>(bufA, bufC, NN * 256);

  // layer 3: 256 -> 512
  k_pack_w<<<cdiv(256 * 512, T), T, 0, stream>>>(W3, wpack, 256, 512);
  k_gemm<256, 512><<<gemm_grid, T, 0, stream>>>(bufC, wpack, bufB, NN);
  k_preinit<512><<<cdiv((long)NN * 512, T), T, 0, stream>>>(bufB, dis, b3, bufA);
  k_scatter<512><<<cdiv((long)EE * 128, T), T, 0, stream>>>(ei, dis, bufB, bufA);
  k_lrelu_bf<<<cdiv((long)NN * 512, T), T, 0, stream>>>(bufA, bufC, NN * 512);

  // layer 4: 512 -> 64
  k_pack_w<<<cdiv(512 * 64, T), T, 0, stream>>>(W4, wpack, 512, 64);
  k_gemm<512, 64><<<gemm_grid, T, 0, stream>>>(bufC, wpack, bufB, NN);
  k_preinit<64><<<cdiv((long)NN * 64, T), T, 0, stream>>>(bufB, dis, b4, bufA);
  k_scatter<64><<<cdiv((long)EE * 16, T), T, 0, stream>>>(ei, dis, bufB, bufA);
  k_lrelu_bf<<<cdiv((long)NN * 64, T), T, 0, stream>>>(bufA, bufC, NN * 64);

  // LayerNorm(h4 + residual), pool, head
  k_ln_res<<<cdiv(NN, 8), T, 0, stream>>>(bufA, resid, br, lng, lnb, bufB);
  k_pool_zero<<<cdiv(GG * 64, T), T, 0, stream>>>(poolu);
  k_pool_max<<<cdiv((long)NN * 64, T), T, 0, stream>>>(bufB, batch, poolu);
  k_head<<<1, 64, 0, stream>>>(poolu, f1W, f1b, fng, fnb, f2W, f2b, out);
}
